// Attn_2327872274604
// MI455X (gfx1250) — matrix operand order repaired and checked by reading, not compiled, not run on hardware
//
#include <hip/hip_runtime.h>
#include <stdint.h>

#define HID   512
#define OH    64
#define DDIM  (HID + OH)        // 576
#define FANIN (2*HID + OH)      // 1088
#define BATCH 128
#define LPOS  1024

#define KT    18                // 576 / 32 k-steps
#define NT    32                // 512 / 16 n-tiles
#define WAVES 4
#define THREADS (WAVES*32)      // 128
#define ROWS  (WAVES*16)        // 64 l-rows per workgroup

#define A_DW  (WAVES*KT*32*8)   // 18432 dwords (72 KB)  A fragments, bf16 pairs
#define B_DW  (KT*32*8)         // 4608 dwords  (18 KB)  B fragments for one n-tile
#define SMEM_BYTES ((A_DW + B_DW)*4)   // 92160 B -> 3 workgroups per 320 KB WGP

typedef __attribute__((ext_vector_type(16))) __bf16 v16bf;
typedef __attribute__((ext_vector_type(8)))  float  v8f;

union Frag16 { uint4 q[2]; v16bf v; };

__device__ __forceinline__ unsigned int f2bf(float f) {
  unsigned int u = __float_as_uint(f);
  unsigned int r = u + 0x7fffu + ((u >> 16) & 1u);   // round-to-nearest-even
  return r >> 16;
}
__device__ __forceinline__ uint32_t pack2bf(float lo, float hi) {
  return f2bf(lo) | (f2bf(hi) << 16);
}

// ---------------------------------------------------------------------------
// Kernel 1: part_h[b][h] = bias[h] + sum_d W[h][d] * hidden[b][d]   (tiny)
// ---------------------------------------------------------------------------
__global__ __launch_bounds__(256) void parth_kernel(
    const float* __restrict__ hidden, const float* __restrict__ W,
    const float* __restrict__ bias, float* __restrict__ ph) {
  int b = blockIdx.x;
  const float* hrow = hidden + (size_t)b * HID;
  for (int h = threadIdx.x; h < HID; h += 256) {
    const float* wrow = W + (size_t)h * FANIN;
    float acc = bias[h];
#pragma unroll 8
    for (int d = 0; d < HID; ++d) acc = fmaf(wrow[d], hrow[d], acc);
    ph[(size_t)b * HID + h] = acc;
  }
}

// ---------------------------------------------------------------------------
// Kernel 2: scores[b][l] = sum_h v[h] * tanh( (enc[l,b,:] . We[h,:]) + ph[b][h] )
// One workgroup = (batch b, 64 l-rows). BF16 WMMA, f32 accumulate.
// Double-buffered fragment loads; 90 KB LDS -> 3 concurrent WGs per WGP.
// ---------------------------------------------------------------------------
__global__ __launch_bounds__(THREADS) void attn_energy_kernel(
    const float* __restrict__ enc,    // [L][B][DDIM]
    const float* __restrict__ W,      // [HID][FANIN], We = cols 512..1087
    const float* __restrict__ vvec,   // [HID]
    const float* __restrict__ ph,     // [B][HID]  (includes bias)
    float* __restrict__ scores)       // [B][L] -> d_out
{
  extern __shared__ uint32_t smem[];
  uint32_t* ldsA = smem;          // A fragments, whole K, all 4 waves
  uint32_t* ldsB = smem + A_DW;   // B fragments for current n-tile

  const int b    = blockIdx.x;
  const int l0   = blockIdx.y * ROWS;
  const int tid  = threadIdx.x;
  const int w    = tid >> 5;
  const int lane = tid & 31;
  const int m    = lane & 15;
  const int khalf= lane >> 4;

  // ---- Stage A once: enc rows l0..l0+63, f32 -> bf16, ISA A-fragment layout.
  // A 16x32 bf16: lanes 0-15 hold K{0..7,16..23}, lanes 16-31 hold K{8..15,24..31}.
#pragma unroll 4
  for (int i = 0; i < A_DW/THREADS; ++i) {
    int idx = i*THREADS + tid;         // bf16-pair index
    int row = idx / 288;               // 0..63  (288 f32-pairs per row)
    int kp  = idx % 288;
    float2 f = *(const float2*)(enc + ((size_t)(l0+row)*BATCH + b)*DDIM + 2*kp);
    uint32_t pk = pack2bf(f.x, f.y);
    int kt = kp >> 4;
    int kr = kp & 15;                  // pair index within the 32-wide k-step
    int kh   = (kr >> 2) & 1;          // which lane-half owns this K pair
    int part = kr >> 3;                // VGPR group 0..3 vs 4..7
    int j    = kr & 3;
    int dst = ((((row >> 4)*KT + kt)*32) + (kh*16 + (row & 15)))*8 + part*4 + j;
    ldsA[dst] = pk;
  }
  __syncthreads();

  float rowacc[8];
#pragma unroll
  for (int r = 0; r < 8; ++r) rowacc[r] = 0.0f;

  const uint4* qa0 = (const uint4*)ldsA + (size_t)(w*KT)*32*2 + lane*2;
  const uint4* qb0 = (const uint4*)ldsB + lane*2;

  for (int nt = 0; nt < NT; ++nt) {
    const int n0 = nt * 16;

    // ---- Stage B strip: We rows n0..n0+15, all 576 d, B-fragment layout.
    // B 32x16 bf16: lanes 0-15 hold K{0..7,16..23}, lanes 16-31 hold K{8..15,24..31}.
#pragma unroll
    for (int i = 0; i < B_DW/THREADS; ++i) {
      int idx = i*THREADS + tid;
      int n  = idx / 288;
      int kp = idx % 288;
      float2 f = *(const float2*)(W + (size_t)(n0+n)*FANIN + HID + 2*kp);
      uint32_t pk = pack2bf(f.x, f.y);
      int kt = kp >> 4;
      int kr = kp & 15;
      int kh = (kr >> 2) & 1;
      int vg = (kr >> 3)*4 + (kr & 3);
      ldsB[(kt*32 + kh*16 + n)*8 + vg] = pk;
    }
    __syncthreads();

    const float phl = ph[(size_t)b*HID + n0 + m];   // per-column (h) bias term
    const float vl  = vvec[n0 + m];

    // ---- K loop, double-buffered fragments (prefetch kt+1 during wmma kt)
    Frag16 fa[2], fb[2];
    fa[0].q[0] = qa0[0];  fa[0].q[1] = qa0[1];
    fb[0].q[0] = qb0[0];  fb[0].q[1] = qb0[1];

    v8f c = {};
#pragma unroll
    for (int kt = 0; kt < KT; ++kt) {
      const int cur = kt & 1, nxt = cur ^ 1;
      if (kt + 1 < KT) {
        const uint4* qa = qa0 + (size_t)(kt+1)*64;   // 32 lanes * 2 uint4
        const uint4* qb = qb0 + (size_t)(kt+1)*64;
        fa[nxt].q[0] = qa[0];  fa[nxt].q[1] = qa[1];
        fb[nxt].q[0] = qb[0];  fb[nxt].q[1] = qb[1];
      }
      c = __builtin_amdgcn_wmma_f32_16x16x32_bf16(
              false, fa[cur].v, false, fb[cur].v, (short)0, c, false, false);
    }

    // energy -> tanh -> weight by v[h], accumulate per-row partial score
#pragma unroll
    for (int r = 0; r < 8; ++r)
      rowacc[r] = fmaf(vl, tanhf(c[r] + phl), rowacc[r]);

    __syncthreads();   // before next n-tile overwrites ldsB
  }

  // Reduce over the 16 columns held by each lane-half (rows m = 8*khalf + r).
#pragma unroll
  for (int r = 0; r < 8; ++r) {
    float s = rowacc[r];
    s += __shfl_xor(s, 1, 32);
    s += __shfl_xor(s, 2, 32);
    s += __shfl_xor(s, 4, 32);
    s += __shfl_xor(s, 8, 32);
    rowacc[r] = s;
  }
  if (m == 0) {
    int lrow = l0 + w*16 + khalf*8;
#pragma unroll
    for (int r = 0; r < 8; ++r)
      scores[(size_t)b*LPOS + lrow + r] = rowacc[r];
  }
}

// ---------------------------------------------------------------------------
// Kernel 3: in-place softmax over L per batch row of d_out.
// ---------------------------------------------------------------------------
__global__ __launch_bounds__(256) void softmax_kernel(float* __restrict__ out) {
  __shared__ float red[256];
  int b = blockIdx.x, t = threadIdx.x;
  float* row = out + (size_t)b * LPOS;
  float x[4];
  float mx = -INFINITY;
#pragma unroll
  for (int i = 0; i < 4; ++i) { x[i] = row[t + 256*i]; mx = fmaxf(mx, x[i]); }
  red[t] = mx; __syncthreads();
  for (int s = 128; s > 0; s >>= 1) {
    if (t < s) red[t] = fmaxf(red[t], red[t + s]);
    __syncthreads();
  }
  mx = red[0]; __syncthreads();
  float sum = 0.0f;
#pragma unroll
  for (int i = 0; i < 4; ++i) { x[i] = expf(x[i] - mx); sum += x[i]; }
  red[t] = sum; __syncthreads();
  for (int s = 128; s > 0; s >>= 1) {
    if (t < s) red[t] += red[t + s];
    __syncthreads();
  }
  float inv = 1.0f / red[0];
#pragma unroll
  for (int i = 0; i < 4; ++i) row[t + 256*i] = x[i] * inv;
}

// ---------------------------------------------------------------------------
extern "C" void kernel_launch(void* const* d_in, const int* in_sizes, int n_in,
                              void* d_out, int out_size, void* d_ws, size_t ws_size,
                              hipStream_t stream) {
  const float* hidden = (const float*)d_in[0];   // [128][512]
  const float* enc    = (const float*)d_in[1];   // [1024][128][576]
  const float* W      = (const float*)d_in[2];   // [512][1088]
  const float* bias   = (const float*)d_in[3];   // [512]
  const float* v      = (const float*)d_in[4];   // [512]
  float* out = (float*)d_out;                    // [128][1][1024]
  float* ph  = (float*)d_ws;                     // [128][512] scratch

  parth_kernel<<<BATCH, 256, 0, stream>>>(hidden, W, bias, ph);

  (void)hipFuncSetAttribute(reinterpret_cast<const void*>(attn_energy_kernel),
                            hipFuncAttributeMaxDynamicSharedMemorySize, SMEM_BYTES);
  dim3 grid(BATCH, LPOS / ROWS);
  attn_energy_kernel<<<grid, THREADS, SMEM_BYTES, stream>>>(enc, W, v, ph, out);

  softmax_kernel<<<BATCH, 256, 0, stream>>>(out);
}
